// Attention4DDownsample_14078902797037
// MI455X (gfx1250) — compile-verified, matrix operand order checked by reading, not run on hardware
//
#include <hip/hip_runtime.h>
#include <hip/hip_bf16.h>
#include <math.h>

typedef _Float16 v16h __attribute__((ext_vector_type(16)));
typedef _Float16 v8h  __attribute__((ext_vector_type(8)));
typedef float    v8f  __attribute__((ext_vector_type(8)));

#define B_    256
#define DIM_  384
#define RES_  14
#define RES2_ 7
#define N_    196
#define N2_   49
#define H_    8
#define KD_   16
#define D_    64
#define NHKD_ 128
#define DH_   512
#define OUTD_ 768
#define SCALE_ 0.25f

// ---------------------------------------------------------------------------
// Fragment helper: CDNA5 wave32 16-bit A/B fragment layout.
// Lane L (0..15):  row = L,      K = kb..kb+7 and kb+16..kb+23, kb = 0
// Lane L (16..31): row = L-16,   kb = 8
// Source row-major; caller passes p = base + row*stride + kb (16B aligned).
// ---------------------------------------------------------------------------
__device__ __forceinline__ v16h make_frag(const _Float16* p) {
  v8h lo = *(const v8h*)p;
  v8h hi = *(const v8h*)(p + 16);
  v16h a;
#pragma unroll
  for (int i = 0; i < 8; ++i) { a[i] = lo[i]; a[i + 8] = hi[i]; }
  return a;
}

// ---------------------------------------------------------------------------
// Fold BN scale into f16 weight, build combined f32 bias: cb = b*s + bn_b
// ---------------------------------------------------------------------------
__global__ void fold_weights_kernel(const float* __restrict__ w,
                                    const float* __restrict__ b,
                                    const float* __restrict__ s,
                                    const float* __restrict__ bb,
                                    _Float16* __restrict__ w16,
                                    float* __restrict__ cb,
                                    int OC, int IC) {
  int tid = blockIdx.x * blockDim.x + threadIdx.x;
  if (tid < OC * IC) {
    int oc = tid / IC;
    w16[tid] = (_Float16)(w[tid] * s[oc]);
  }
  if (tid < OC) cb[tid] = b[tid] * s[tid] + bb[tid];
}

// Gather attention bias table: biasfull[h][q][n] = attn_biases[h][idxs[q*196+n]]
__global__ void bias_gather_kernel(const float* __restrict__ ab,
                                   const int* __restrict__ idxs,
                                   float* __restrict__ out, int n_off) {
  int tid = blockIdx.x * blockDim.x + threadIdx.x;
  if (tid >= H_ * N2_ * N_) return;
  int h = tid / (N2_ * N_);
  int qn = tid % (N2_ * N_);
  out[tid] = ab[h * n_off + idxs[qn]];
}

// x (B,384,196) f32 -> xT (B,196,384) f16  (spatial-major for B-fragments)
__global__ void transpose_x_kernel(const float* __restrict__ x,
                                   _Float16* __restrict__ xT) {
  size_t tid = (size_t)blockIdx.x * blockDim.x + threadIdx.x;
  if (tid >= (size_t)B_ * N_ * DIM_) return;
  int c = tid % DIM_;
  int s = (tid / DIM_) % N_;
  int b = tid / ((size_t)DIM_ * N_);
  xT[tid] = (_Float16)x[((size_t)b * DIM_ + c) * N_ + s];
}

// q input: depthwise 3x3 stride2 pad1 + bias + stride-2 pool, written as
// qinT (B,49,384) f16
__global__ void qprep_kernel(const float* __restrict__ x,
                             const float* __restrict__ qlw,
                             const float* __restrict__ qlb,
                             _Float16* __restrict__ qinT) {
  size_t tid = (size_t)blockIdx.x * blockDim.x + threadIdx.x;
  if (tid >= (size_t)B_ * N2_ * DIM_) return;
  int c  = tid % DIM_;
  int s2 = (tid / DIM_) % N2_;
  int b  = tid / ((size_t)DIM_ * N2_);
  int oy = s2 / RES2_, ox = s2 % RES2_;
  const float* xb = x + ((size_t)b * DIM_ + c) * N_;
  float acc = xb[(2 * oy) * RES_ + (2 * ox)];  // pool
  float conv = 0.f;
#pragma unroll
  for (int ky = 0; ky < 3; ++ky) {
#pragma unroll
    for (int kx = 0; kx < 3; ++kx) {
      int iy = 2 * oy - 1 + ky, ix = 2 * ox - 1 + kx;
      if (iy >= 0 && iy < RES_ && ix >= 0 && ix < RES_)
        conv += xb[iy * RES_ + ix] * qlw[c * 9 + ky * 3 + kx];
    }
  }
  qinT[tid] = (_Float16)(acc + conv + qlb[c]);
}

// ---------------------------------------------------------------------------
// WMMA GEMM, 32x32 output per wave (2x2 tiles of 16x16): each A/B fragment is
// reused twice -> 4 WMMAs per 4 fragment loads per k-step.
// W: [OC][IC] f16 (BN-scaled), X: [B][S][IC] f16.
// OC multiple of 32; IC multiple of 32; S guarded (rows clamped for loads).
// ---------------------------------------------------------------------------
template <int OUT_F32>
__global__ __launch_bounds__(32)
void gemm32_kernel(const _Float16* __restrict__ W,
                   const float* __restrict__ cb,
                   const _Float16* __restrict__ X,
                   int S, int IC, int OC,
                   void* __restrict__ out) {
  int lane = threadIdx.x;
  int st  = blockIdx.x * 32;
  int oc0 = blockIdx.y * 32;
  int b   = blockIdx.z;
  int r  = lane & 15;
  int kb = (lane >> 4) * 8;
  int s0 = st + r;        if (s0 >= S) s0 = S - 1;   // clamp, guarded at store
  int s1 = st + 16 + r;   if (s1 >= S) s1 = S - 1;
  const _Float16* wrow0 = W + (size_t)(oc0 + r) * IC + kb;
  const _Float16* wrow1 = W + (size_t)(oc0 + 16 + r) * IC + kb;
  const _Float16* xrow0 = X + ((size_t)b * S + s0) * IC + kb;
  const _Float16* xrow1 = X + ((size_t)b * S + s1) * IC + kb;
  v8f acc00 = {}, acc01 = {}, acc10 = {}, acc11 = {};
  for (int k0 = 0; k0 < IC; k0 += 32) {
    if (k0 + 32 < IC) {  // lowers to global_prefetch_b8
      __builtin_prefetch(wrow0 + k0 + 32, 0, 3);
      __builtin_prefetch(wrow1 + k0 + 32, 0, 3);
      __builtin_prefetch(xrow0 + k0 + 32, 0, 3);
      __builtin_prefetch(xrow1 + k0 + 32, 0, 3);
    }
    v16h a0 = make_frag(wrow0 + k0);
    v16h a1 = make_frag(wrow1 + k0);
    v16h b0 = make_frag(xrow0 + k0);
    v16h b1 = make_frag(xrow1 + k0);
    acc00 = __builtin_amdgcn_wmma_f32_16x16x32_f16(false, a0, false, b0,
                                                   (short)0, acc00, false, false);
    acc01 = __builtin_amdgcn_wmma_f32_16x16x32_f16(false, a0, false, b1,
                                                   (short)0, acc01, false, false);
    acc10 = __builtin_amdgcn_wmma_f32_16x16x32_f16(false, a1, false, b0,
                                                   (short)0, acc10, false, false);
    acc11 = __builtin_amdgcn_wmma_f32_16x16x32_f16(false, a1, false, b1,
                                                   (short)0, acc11, false, false);
  }
  int mbase = (lane >> 4) * 8;
  int n = lane & 15;
#pragma unroll
  for (int mt = 0; mt < 2; ++mt) {
#pragma unroll
    for (int nt = 0; nt < 2; ++nt) {
      const v8f& acc = (mt == 0) ? (nt == 0 ? acc00 : acc01)
                                 : (nt == 0 ? acc10 : acc11);
      int s = st + nt * 16 + n;
      if (s < S) {
#pragma unroll
        for (int j = 0; j < 8; ++j) {
          int oc = oc0 + mt * 16 + mbase + j;
          float y = acc[j] + cb[oc];
          if (OUT_F32)
            ((float*)out)[((size_t)b * OC + oc) * S + s] = y;
          else
            ((_Float16*)out)[((size_t)b * OC + oc) * S + s] = (_Float16)y;
        }
      }
    }
  }
}

// ---------------------------------------------------------------------------
// Attention per (head, batch): q(49x16) kT(16x196) v(196x64), softmax + bias.
// 4 waves, wave w owns query rows [16w, 16w+16). Logits kept in registers,
// softmax via __shfl_xor within 16-lane half-groups (C-fragment layout),
// probabilities staged in LDS f16 for the AV WMMAs.
// Bias loads are unconditional with clamped indices (padding rows/cols are
// don't-care) -> no per-element EXEC branching.
// LDS phases: [Q(64x32) | Kt(208x32)] reused for V(64x224); P(64x224) resident.
// ---------------------------------------------------------------------------
__global__ __launch_bounds__(128)
void attn_kernel(const _Float16* __restrict__ q16,
                 const _Float16* __restrict__ k16,
                 const _Float16* __restrict__ v16,
                 const float* __restrict__ biasfull,
                 float* __restrict__ o) {
  __shared__ _Float16 smem[14336 * 2];  // 57344 B
  _Float16* sQ = smem;            // [64][32]
  _Float16* sK = smem + 2048;     // [208][32]  (n-major, kd padded to 32)
  _Float16* sV = smem;            // [64][224]  (phase 2, overlaps sQ/sK)
  _Float16* sP = smem + 14336;    // [64][224]

  int tid = threadIdx.x;
  int h = blockIdx.x, b = blockIdx.y;
  const _Float16* qb = q16 + ((size_t)b * NHKD_ + h * KD_) * N2_;
  const _Float16* kb_ = k16 + ((size_t)b * NHKD_ + h * KD_) * N_;
  const _Float16* vb = v16 + ((size_t)b * DH_ + h * D_) * N_;
  const float* bias = biasfull + (size_t)h * N2_ * N_;

  // zero only the pad columns of sP (208..223); softmax writes the rest
  for (int i = tid; i < 64 * 16; i += 128) {
    int row = i >> 4, col = 208 + (i & 15);
    sP[row * 224 + col] = (_Float16)0.f;
  }
  for (int i = tid; i < 64 * 32; i += 128) {
    int m = i >> 5, kd = i & 31;
    sQ[i] = (m < N2_ && kd < KD_) ? qb[kd * N2_ + m] : (_Float16)0.f;
  }
  for (int i = tid; i < 208 * 32; i += 128) {
    int n = i >> 5, kd = i & 31;
    sK[i] = (n < N_ && kd < KD_) ? kb_[kd * N_ + n] : (_Float16)0.f;
  }
  __syncthreads();

  int wid = tid >> 5, lane = tid & 31;
  int r = lane & 15, kb = (lane >> 4) * 8;
  int mrow = (lane >> 4) * 8;

  // per-j clamped bias row offsets (rows >= 49 are don't-care)
  int boff[8];
#pragma unroll
  for (int j = 0; j < 8; ++j) {
    int q = wid * 16 + mrow + j;
    boff[j] = ((q < N2_) ? q : (N2_ - 1)) * N_;
  }

  v16h aq = make_frag(sQ + (wid * 16 + r) * 32 + kb);

  float l[13][8];
#pragma unroll
  for (int t = 0; t < 13; ++t) {
    v16h bf = make_frag(sK + (t * 16 + r) * 32 + kb);
    v8f c = {};
    c = __builtin_amdgcn_wmma_f32_16x16x32_f16(false, aq, false, bf,
                                               (short)0, c, false, false);
    int n = t * 16 + (lane & 15);
    int nc = (n < N_) ? n : (N_ - 1);  // clamp, cols >= 196 are masked below
    bool valid = (n < N_);
#pragma unroll
    for (int j = 0; j < 8; ++j) {
      float bv = bias[boff[j] + nc];   // unconditional load
      l[t][j] = valid ? (c[j] * SCALE_ + bv) : -1e30f;
    }
  }
  __syncthreads();  // done reading sQ/sK before they are overwritten by sV

  // register softmax over n (13 tiles x 16 lanes of the half-group)
  float mx[8], sum[8];
#pragma unroll
  for (int j = 0; j < 8; ++j) mx[j] = -1e30f;
#pragma unroll
  for (int t = 0; t < 13; ++t)
#pragma unroll
    for (int j = 0; j < 8; ++j) mx[j] = fmaxf(mx[j], l[t][j]);
#pragma unroll
  for (int j = 0; j < 8; ++j) {
#pragma unroll
    for (int m = 1; m < 16; m <<= 1)
      mx[j] = fmaxf(mx[j], __shfl_xor(mx[j], m, 32));
  }
#pragma unroll
  for (int j = 0; j < 8; ++j) sum[j] = 0.f;
#pragma unroll
  for (int t = 0; t < 13; ++t)
#pragma unroll
    for (int j = 0; j < 8; ++j) {
      float e = expf(l[t][j] - mx[j]);
      l[t][j] = e;
      sum[j] += e;
    }
#pragma unroll
  for (int j = 0; j < 8; ++j) {
#pragma unroll
    for (int m = 1; m < 16; m <<= 1) sum[j] += __shfl_xor(sum[j], m, 32);
    sum[j] = 1.f / sum[j];
  }
#pragma unroll
  for (int t = 0; t < 13; ++t) {
    int n = t * 16 + (lane & 15);
#pragma unroll
    for (int j = 0; j < 8; ++j) {
      int row = wid * 16 + mrow + j;
      sP[row * 224 + n] = (_Float16)(l[t][j] * sum[j]);
    }
  }
  // load V^T tiles: sV[d][s1], s1 padded 196->224
  for (int i = tid; i < 64 * 224; i += 128) {
    int d = i / 224, s1 = i % 224;
    sV[i] = (s1 < N_) ? vb[d * N_ + s1] : (_Float16)0.f;
  }
  __syncthreads();

  // o = P(49x224) @ V(224x64)
  v8f oacc[4] = {{}, {}, {}, {}};
  for (int ks = 0; ks < 7; ++ks) {
    v16h ap = make_frag(sP + (wid * 16 + r) * 224 + ks * 32 + kb);
#pragma unroll
    for (int nt = 0; nt < 4; ++nt) {
      v16h bv = make_frag(sV + (nt * 16 + r) * 224 + ks * 32 + kb);
      oacc[nt] = __builtin_amdgcn_wmma_f32_16x16x32_f16(false, ap, false, bv,
                                                        (short)0, oacc[nt],
                                                        false, false);
    }
  }
  int n = lane & 15;
#pragma unroll
  for (int nt = 0; nt < 4; ++nt) {
#pragma unroll
    for (int j = 0; j < 8; ++j) {
      int q = wid * 16 + mrow + j;
      if (q < N2_)
        o[((size_t)b * DH_ + h * D_ + nt * 16 + n) * N2_ + q] = oacc[nt][j];
    }
  }
}

// ---------------------------------------------------------------------------
// v_local depthwise conv + BN + add attention output + exact GELU,
// written transposed as g16 (B,49,512) f16 for the proj GEMM.
// ---------------------------------------------------------------------------
__global__ void fuse_kernel(const float* __restrict__ o,
                            const _Float16* __restrict__ v16,
                            const float* __restrict__ vlw,
                            const float* __restrict__ vlb,
                            const float* __restrict__ vls,
                            const float* __restrict__ vlbb,
                            _Float16* __restrict__ g16) {
  size_t tid = (size_t)blockIdx.x * blockDim.x + threadIdx.x;
  if (tid >= (size_t)B_ * N2_ * DH_) return;
  int c  = tid % DH_;
  int s2 = (tid / DH_) % N2_;
  int b  = tid / ((size_t)DH_ * N2_);
  int oy = s2 / RES2_, ox = s2 % RES2_;
  const _Float16* vch = v16 + ((size_t)b * DH_ + c) * N_;
  float conv = 0.f;
#pragma unroll
  for (int ky = 0; ky < 3; ++ky) {
#pragma unroll
    for (int kx = 0; kx < 3; ++kx) {
      int iy = 2 * oy - 1 + ky, ix = 2 * ox - 1 + kx;
      if (iy >= 0 && iy < RES_ && ix >= 0 && ix < RES_)
        conv += (float)vch[iy * RES_ + ix] * vlw[c * 9 + ky * 3 + kx];
    }
  }
  float vl = (conv + vlb[c]) * vls[c] + vlbb[c];
  float t = o[((size_t)b * DH_ + c) * N2_ + s2] + vl;
  float g = 0.5f * t * (1.f + erff(t * 0.70710678118654752f));
  g16[tid] = (_Float16)g;
}

// ---------------------------------------------------------------------------
extern "C" void kernel_launch(void* const* d_in, const int* in_sizes, int n_in,
                              void* d_out, int out_size, void* d_ws, size_t ws_size,
                              hipStream_t stream) {
  (void)n_in; (void)out_size; (void)ws_size;
  const float* x      = (const float*)d_in[0];
  const float* qlw    = (const float*)d_in[1];
  const float* qlb    = (const float*)d_in[2];
  const float* qpw    = (const float*)d_in[3];
  const float* qpb    = (const float*)d_in[4];
  const float* qbs    = (const float*)d_in[5];
  const float* qbb    = (const float*)d_in[6];
  const float* kw     = (const float*)d_in[7];
  const float* kbi    = (const float*)d_in[8];
  const float* kbs    = (const float*)d_in[9];
  const float* kbb    = (const float*)d_in[10];
  const float* vw     = (const float*)d_in[11];
  const float* vbi    = (const float*)d_in[12];
  const float* vbs    = (const float*)d_in[13];
  const float* vbb    = (const float*)d_in[14];
  const float* vlw    = (const float*)d_in[15];
  const float* vlb    = (const float*)d_in[16];
  const float* vls    = (const float*)d_in[17];
  const float* vlbb   = (const float*)d_in[18];
  const float* pw     = (const float*)d_in[19];
  const float* pb     = (const float*)d_in[20];
  const float* pbs    = (const float*)d_in[21];
  const float* pbb    = (const float*)d_in[22];
  const float* abias  = (const float*)d_in[23];
  const int*   idxs   = (const int*)d_in[24];
  int n_off = in_sizes[23] / H_;

  char* p = (char*)d_ws;
  auto alloc = [&](size_t bytes) {
    void* r = (void*)p;
    p += (bytes + 255) & ~(size_t)255;
    return r;
  };
  _Float16* xT   = (_Float16*)alloc((size_t)B_ * N_ * DIM_ * 2);
  _Float16* qinT = (_Float16*)alloc((size_t)B_ * N2_ * DIM_ * 2);
  _Float16* wk16 = (_Float16*)alloc((size_t)NHKD_ * DIM_ * 2);
  _Float16* wv16 = (_Float16*)alloc((size_t)DH_ * DIM_ * 2);
  _Float16* wq16 = (_Float16*)alloc((size_t)NHKD_ * DIM_ * 2);
  _Float16* wp16 = (_Float16*)alloc((size_t)OUTD_ * DH_ * 2);
  float* cbk = (float*)alloc(NHKD_ * 4);
  float* cbv = (float*)alloc(DH_ * 4);
  float* cbq = (float*)alloc(NHKD_ * 4);
  float* cbp = (float*)alloc(OUTD_ * 4);
  float* biasfull = (float*)alloc((size_t)H_ * N2_ * N_ * 4);
  _Float16* k16 = (_Float16*)alloc((size_t)B_ * NHKD_ * N_ * 2);
  _Float16* v16 = (_Float16*)alloc((size_t)B_ * DH_ * N_ * 2);
  _Float16* q16 = (_Float16*)alloc((size_t)B_ * NHKD_ * N2_ * 2);
  float*    obuf = (float*)alloc((size_t)B_ * DH_ * N2_ * 4);
  _Float16* g16 = (_Float16*)alloc((size_t)B_ * N2_ * DH_ * 2);

  // 1) fold BN into weights
  fold_weights_kernel<<<(NHKD_ * DIM_ + 255) / 256, 256, 0, stream>>>(
      kw, kbi, kbs, kbb, wk16, cbk, NHKD_, DIM_);
  fold_weights_kernel<<<(DH_ * DIM_ + 255) / 256, 256, 0, stream>>>(
      vw, vbi, vbs, vbb, wv16, cbv, DH_, DIM_);
  fold_weights_kernel<<<(NHKD_ * DIM_ + 255) / 256, 256, 0, stream>>>(
      qpw, qpb, qbs, qbb, wq16, cbq, NHKD_, DIM_);
  fold_weights_kernel<<<(OUTD_ * DH_ + 255) / 256, 256, 0, stream>>>(
      pw, pb, pbs, pbb, wp16, cbp, OUTD_, DH_);

  // 2) bias table gather
  bias_gather_kernel<<<(H_ * N2_ * N_ + 255) / 256, 256, 0, stream>>>(
      abias, idxs, biasfull, n_off);

  // 3) activation prep (f16, spatial-major)
  {
    size_t n = (size_t)B_ * N_ * DIM_;
    transpose_x_kernel<<<(unsigned)((n + 255) / 256), 256, 0, stream>>>(x, xT);
  }
  {
    size_t n = (size_t)B_ * N2_ * DIM_;
    qprep_kernel<<<(unsigned)((n + 255) / 256), 256, 0, stream>>>(x, qlw, qlb, qinT);
  }

  // 4) WMMA GEMM projections (k, v, q): 32x32 per wave
  gemm32_kernel<0><<<dim3(7, NHKD_ / 32, B_), 32, 0, stream>>>(
      wk16, cbk, xT, N_, DIM_, NHKD_, (void*)k16);
  gemm32_kernel<0><<<dim3(7, DH_ / 32, B_), 32, 0, stream>>>(
      wv16, cbv, xT, N_, DIM_, DH_, (void*)v16);
  gemm32_kernel<0><<<dim3(2, NHKD_ / 32, B_), 32, 0, stream>>>(
      wq16, cbq, qinT, N2_, DIM_, NHKD_, (void*)q16);

  // 5) attention (QK^T + bias, softmax, @V), per (head, batch)
  attn_kernel<<<dim3(H_, B_), 128, 0, stream>>>(q16, k16, v16, biasfull, obuf);

  // 6) v_local dwconv + add + GELU -> transposed f16
  {
    size_t n = (size_t)B_ * N2_ * DH_;
    fuse_kernel<<<(unsigned)((n + 255) / 256), 256, 0, stream>>>(
        obuf, v16, vlw, vlb, vls, vlbb, g16);
  }

  // 7) output projection, f32 to d_out (B,768,7,7)
  gemm32_kernel<1><<<dim3(2, OUTD_ / 32, B_), 32, 0, stream>>>(
      wp16, cbp, g16, N2_, DH_, OUTD_, d_out);
}
